// Attention_70832600646207
// MI455X (gfx1250) — compile-verified, matrix operand order checked by reading, not run on hardware
//
#include <hip/hip_runtime.h>
#include <hip/hip_bf16.h>
#include <math.h>

// ---------------- problem constants ----------------
#define BB      2
#define TT      2048
#define DD      3584
#define NH      16          // query heads
#define KVH     8           // kv heads
#define HD      256         // head dim
#define GG      (NH / KVH)  // 2
#define WIN     1024
#define SCALE_F 0.0625f
#define CAP_F   50.0f

typedef __bf16 bf16_t;
typedef __attribute__((ext_vector_type(16))) __bf16 v16bf;
typedef __attribute__((ext_vector_type(8)))  float  v8f;

// -------- gfx1250 async global->LDS staging (ASYNCcnt) via inline asm --------
// VDST VGPR carries the LDS byte address (low 32 bits of the generic pointer,
// per the ISA flat->LDS truncation rule); VADDR is the 64-bit global address.
__device__ __forceinline__ void async_copy_b128(const void* g, void* l) {
    unsigned lds = (unsigned)(unsigned long long)l;
    asm volatile("global_load_async_to_lds_b128 %0, %1, off"
                 :: "v"(lds), "v"(g) : "memory");
}

__device__ __forceinline__ void wait_async() {
    asm volatile("s_wait_asynccnt 0x0" ::: "memory");
}

// ---------------- WMMA helpers (CDNA5 16-bit layouts) ----------------
// K index pattern inside a 16-bit A/B fragment:
//   VGPR j<4  -> K = {2j, 2j+1}        (+8 for lanes 16..31)
//   VGPR j>=4 -> K = {16+2(j-4), +1}   (+8 for lanes 16..31)
// Per lane-half this is two contiguous 8-element (16 B) runs -> ds_load_b128.
__device__ __forceinline__ int kpat(int j, int hi, int half) {
    return ((j < 4) ? (2 * j) : (16 + 2 * (j - 4))) + half * 8 + hi;
}

__device__ __forceinline__ v8f zero8() {
    v8f z;
#pragma unroll
    for (int i = 0; i < 8; ++i) z[i] = 0.0f;
    return z;
}

// A fragment (16x32, MxK), row-major source: s[m*stride + k]
__device__ __forceinline__ v16bf load_A_frag(const bf16_t* s, int stride) {
    const int lane = threadIdx.x & 31, m = lane & 15, half = lane >> 4;
    v16bf a;
#pragma unroll
    for (int j = 0; j < 8; ++j)
#pragma unroll
        for (int hi = 0; hi < 2; ++hi)
            a[2 * j + hi] = s[m * stride + kpat(j, hi, half)];
    return a;
}

// B fragment (32x16, KxN) where B(k,n) = s[n*stride + k]
// (source tile stored N-major so each lane reads two contiguous 16 B runs)
__device__ __forceinline__ v16bf load_B_frag_T(const bf16_t* s, int stride) {
    const int lane = threadIdx.x & 31, n = lane & 15, half = lane >> 4;
    v16bf b;
#pragma unroll
    for (int j = 0; j < 8; ++j)
#pragma unroll
        for (int hi = 0; hi < 2; ++hi)
            b[2 * j + hi] = s[n * stride + kpat(j, hi, half)];
    return b;
}

__device__ __forceinline__ v8f wmma_bf16(v16bf a, v16bf b, v8f c) {
    return __builtin_amdgcn_wmma_f32_16x16x32_bf16(false, a, false, b,
                                                   (short)0, c, false, false);
}

// ---------------- Kernel 1: QKV projection (f32 x f32 -> bf16, bf16 WMMA) ----
// Y[row, head, h] = sum_d X[row, d] * W[head, d, h]
// grid: (M/16, HD/64, heads); block 128 (4 waves, each owns a 16x16 tile)
__global__ __launch_bounds__(128) void proj_gemm_kernel(
    const float* __restrict__ X,   // [M, D]
    const float* __restrict__ W,   // [heads, D, HD]
    bf16_t* __restrict__ Y,        // [M, heads*HD]
    int M, int Dk, int heads)
{
    __shared__ alignas(16) bf16_t As[16][40];   // 16 rows x 32 K (pad 40)
    __shared__ alignas(16) bf16_t Bs[64][40];   // N-major: [n][k], 64 x 32 (pad 40)

    const int tid  = threadIdx.x;
    const int w    = tid >> 5;
    const int lane = tid & 31, ln = lane & 15, half = lane >> 4;
    const int m0   = blockIdx.x * 16;
    const int n0wg = blockIdx.y * 64;
    const int head = blockIdx.z;
    const size_t wbase = (size_t)head * Dk * HD;

    v8f c = zero8();
    for (int kk = 0; kk < Dk; kk += 32) {
        // stage A (16x32 f32 -> bf16), coalesced reads
#pragma unroll
        for (int u = 0; u < 4; ++u) {
            int e = tid + u * 128, r = e >> 5, cc = e & 31;
            As[r][cc] = (bf16_t)X[(size_t)(m0 + r) * Dk + kk + cc];
        }
        // stage B transposed (32 K-rows x 64 N-cols -> Bs[n][k])
#pragma unroll
        for (int u = 0; u < 16; ++u) {
            int e = tid + u * 128, r = e >> 6, cc = e & 63;
            Bs[cc][r] = (bf16_t)W[wbase + (size_t)(kk + r) * HD + n0wg + cc];
        }
        __syncthreads();
        v16bf a = load_A_frag(&As[0][0], 40);
        v16bf b = load_B_frag_T(&Bs[w * 16][0], 40);
        c = wmma_bf16(a, b, c);
        __syncthreads();
    }
    const int n0 = n0wg + w * 16;
#pragma unroll
    for (int i = 0; i < 8; ++i) {
        int row = m0 + i + 8 * half;
        Y[(size_t)row * heads * HD + (size_t)head * HD + n0 + ln] = (bf16_t)c[i];
    }
}

// ---------------- Kernel 2: RoPE (in-place on bf16 q and k; q also * SCALE) ----
__global__ __launch_bounds__(256) void rope_kernel(
    bf16_t* __restrict__ q, bf16_t* __restrict__ k,
    const int* __restrict__ segpos)
{
    const int QP = BB * TT * NH * (HD / 2);
    const int KP = BB * TT * KVH * (HD / 2);
    int i = blockIdx.x * blockDim.x + threadIdx.x;
    const float lc = 9.210340371976184f / 128.0f;  // log(10000) / (H/2)

    if (i < QP) {
        int h = i & 127, r = i >> 7;
        int n = r % NH; r /= NH;
        int t = r % TT; int b = r / TT;
        size_t base = ((size_t)(b * TT + t) * NH + n) * HD;
        float pos = (float)segpos[b * TT + t];
        float inv = __expf(-(float)h * lc);
        float s, co; __sincosf(pos * inv, &s, &co);
        float x1 = (float)q[base + h], x2 = (float)q[base + h + 128];
        q[base + h]       = (bf16_t)((x1 * co - x2 * s) * SCALE_F);
        q[base + h + 128] = (bf16_t)((x2 * co + x1 * s) * SCALE_F);
    } else if (i < QP + KP) {
        int j = i - QP;
        int h = j & 127, r = j >> 7;
        int kh = r % KVH; r /= KVH;
        int t = r % TT; int b = r / TT;
        size_t base = ((size_t)(b * TT + t) * KVH + kh) * HD;
        float pos = (float)segpos[b * TT + t];
        float inv = __expf(-(float)h * lc);
        float s, co; __sincosf(pos * inv, &s, &co);
        float x1 = (float)k[base + h], x2 = (float)k[base + h + 128];
        k[base + h]       = (bf16_t)(x1 * co - x2 * s);
        k[base + h + 128] = (bf16_t)(x2 * co + x1 * s);
    }
}

// ---------------- Kernel 3: sliding-window flash attention ----------------
// grid: (T/64, NH, B); block 128. Wave w handles query rows [m0+16w, m0+16w+16).
// Q tile lives in registers (8 A-frags). K/V tiles of 32 keys stream through LDS:
// K row-major (async global->LDS), V transposed (for contiguous B-frag loads).
__global__ __launch_bounds__(128) void attn_kernel(
    const bf16_t* __restrict__ qb,  // [B,T,NH,HD]
    const bf16_t* __restrict__ kb,  // [B,T,KVH,HD]
    const bf16_t* __restrict__ vb,  // [B,T,KVH,HD]
    bf16_t* __restrict__ enc)       // [B,T,NH,HD]
{
    __shared__ alignas(16) bf16_t k_s[32][264];     // [s][h], pad 264 (528 B rows)
    __shared__ alignas(16) bf16_t v_t[HD][40];      // [h][s], pad 40  (80 B rows)
    __shared__ alignas(16) bf16_t p_s[4][16][40];   // per-wave P tile [m][s]

    const int b = blockIdx.z, n = blockIdx.y;
    const int kh = n / GG;
    const int m0 = blockIdx.x * 64;
    const int tid = threadIdx.x, w = tid >> 5;
    const int lane = tid & 31, ln = lane & 15, half = lane >> 4;
    const int mw = m0 + w * 16;

    // ---- Q tile -> 8 register A-fragments (16 rows x 256 cols bf16) ----
    v16bf qf[8];
    {
        const bf16_t* qbase = qb + ((size_t)(b * TT + mw + ln) * NH + n) * HD;
#pragma unroll
        for (int kk = 0; kk < 8; ++kk)
#pragma unroll
            for (int j = 0; j < 8; ++j)
#pragma unroll
                for (int hi = 0; hi < 2; ++hi)
                    qf[kk][2 * j + hi] = qbase[kk * 32 + kpat(j, hi, half)];
    }

    v8f acc[16];
#pragma unroll
    for (int j = 0; j < 16; ++j) acc[j] = zero8();
    float mrow[8], lrow[8];
#pragma unroll
    for (int i = 0; i < 8; ++i) { mrow[i] = -INFINITY; lrow[i] = 0.0f; }

    int s_begin = m0 - (WIN - 1);
    if (s_begin < 0) s_begin = 0;
    s_begin &= ~31;
    const int s_last = m0 + 63;

    for (int s0 = s_begin; s0 <= s_last; s0 += 32) {
        __syncthreads();
        // ---- K tile 32x256: async global->LDS, 16 B chunks ----
#pragma unroll
        for (int u = 0; u < 8; ++u) {
            int e = tid + u * 128, r = e >> 5, cvec = e & 31;
            const bf16_t* ksrc =
                kb + ((size_t)(b * TT + s0 + r) * KVH + kh) * HD + cvec * 8;
            async_copy_b128(ksrc, &k_s[r][cvec * 8]);
        }
        // ---- V tile transposed: vectorized global read, 2 B LDS scatter ----
#pragma unroll
        for (int u = 0; u < 8; ++u) {
            int e = tid + u * 128, r = e >> 5, hblk = e & 31;  // h0 = hblk*8
            const bf16_t* vsrc =
                vb + ((size_t)(b * TT + s0 + r) * KVH + kh) * HD + hblk * 8;
            uint4 raw = *(const uint4*)vsrc;
            const bf16_t* pv = (const bf16_t*)&raw;
#pragma unroll
            for (int j = 0; j < 8; ++j) v_t[hblk * 8 + j][r] = pv[j];
        }
        wait_async();
        __syncthreads();

        // ---- logits = Q * K^T (two 16-col C-fragments) ----
        v8f c0 = zero8(), c1 = zero8();
#pragma unroll
        for (int kk = 0; kk < 8; ++kk) {
            v16bf b0 = load_B_frag_T(&k_s[0][kk * 32],  264);
            v16bf b1 = load_B_frag_T(&k_s[16][kk * 32], 264);
            c0 = wmma_bf16(qf[kk], b0, c0);
            c1 = wmma_bf16(qf[kk], b1, c1);
        }

        // ---- soft-cap, mask, online softmax (row r <-> VGPR r%8, half r/8) ----
        float alph[8];
#pragma unroll
        for (int i = 0; i < 8; ++i) {
            int srow = mw + i + 8 * half;
            float x0 = tanhf(c0[i] * (1.0f / CAP_F)) * CAP_F;
            float x1 = tanhf(c1[i] * (1.0f / CAP_F)) * CAP_F;
            int col0 = s0 + ln, col1 = col0 + 16;
            x0 = ((col0 <= srow) && (col0 > srow - WIN)) ? x0 : -INFINITY;
            x1 = ((col1 <= srow) && (col1 > srow - WIN)) ? x1 : -INFINITY;
            float tm = fmaxf(x0, x1);
#pragma unroll
            for (int msk = 1; msk < 16; msk <<= 1) tm = fmaxf(tm, __shfl_xor(tm, msk, 32));
            float nm = fmaxf(mrow[i], tm);
            float a  = (nm == -INFINITY) ? 1.0f : __expf(mrow[i] - nm);
            float p0 = (x0 == -INFINITY) ? 0.0f : __expf(x0 - nm);
            float p1 = (x1 == -INFINITY) ? 0.0f : __expf(x1 - nm);
            float ls = p0 + p1;
#pragma unroll
            for (int msk = 1; msk < 16; msk <<= 1) ls += __shfl_xor(ls, msk, 32);
            mrow[i] = nm;
            lrow[i] = lrow[i] * a + ls;
            alph[i] = a;
            p_s[w][i + 8 * half][ln]      = (bf16_t)p0;
            p_s[w][i + 8 * half][ln + 16] = (bf16_t)p1;
        }
#pragma unroll
        for (int j = 0; j < 16; ++j)
#pragma unroll
            for (int i = 0; i < 8; ++i) acc[j][i] *= alph[i];
        __syncthreads();   // P visible across lanes; orders vs next restage

        // ---- acc += P * V (K = 32 keys, full H = 16 col-tiles) ----
        v16bf pa = load_A_frag(&p_s[w][0][0], 40);
#pragma unroll
        for (int j = 0; j < 16; ++j) {
            v16bf vfr = load_B_frag_T(&v_t[j * 16][0], 40);
            acc[j] = wmma_bf16(pa, vfr, acc[j]);
        }
    }

    // ---- normalize + store bf16 ----
#pragma unroll
    for (int j = 0; j < 16; ++j)
#pragma unroll
        for (int i = 0; i < 8; ++i) {
            int row = mw + i + 8 * half;
            float l = lrow[i];
            float o = (l > 0.0f) ? acc[j][i] / l : 0.0f;
            enc[((size_t)(b * TT + row) * NH + n) * HD + j * 16 + ln] = (bf16_t)o;
        }
}

// ---------------- Kernel 4: output projection (bf16 x f32->bf16 -> f32) --------
// Y[row, d] = sum_k A[row, k] * Wo[k, d],  Ka = NH*HD = 4096, Dd = 3584
__global__ __launch_bounds__(128) void outproj_kernel(
    const bf16_t* __restrict__ A,  // [M, Ka]
    const float* __restrict__ Wo,  // [Ka, Dd]
    float* __restrict__ Y,         // [M, Dd]
    int M, int Ka, int Dd)
{
    __shared__ alignas(16) bf16_t As[16][40];
    __shared__ alignas(16) bf16_t Bs[64][40];  // N-major: [n][k]

    const int tid = threadIdx.x, w = tid >> 5;
    const int lane = tid & 31, ln = lane & 15, half = lane >> 4;
    const int m0 = blockIdx.x * 16;
    const int n0wg = blockIdx.y * 64;

    v8f c = zero8();
    for (int kk = 0; kk < Ka; kk += 32) {
#pragma unroll
        for (int u = 0; u < 4; ++u) {
            int e = tid + u * 128, r = e >> 5, cc = e & 31;
            As[r][cc] = A[(size_t)(m0 + r) * Ka + kk + cc];
        }
#pragma unroll
        for (int u = 0; u < 16; ++u) {
            int e = tid + u * 128, r = e >> 6, cc = e & 63;
            Bs[cc][r] = (bf16_t)Wo[(size_t)(kk + r) * Dd + n0wg + cc];
        }
        __syncthreads();
        v16bf a = load_A_frag(&As[0][0], 40);
        v16bf b = load_B_frag_T(&Bs[w * 16][0], 40);
        c = wmma_bf16(a, b, c);
        __syncthreads();
    }
    const int n0 = n0wg + w * 16;
#pragma unroll
    for (int i = 0; i < 8; ++i)
        Y[(size_t)(m0 + i + 8 * half) * Dd + n0 + ln] = c[i];
}

// ---------------- host launcher ----------------
extern "C" void kernel_launch(void* const* d_in, const int* in_sizes, int n_in,
                              void* d_out, int out_size, void* d_ws, size_t ws_size,
                              hipStream_t stream) {
    const float* x      = (const float*)d_in[0];
    const int*   segpos = (const int*)  d_in[1];
    // d_in[2] = attn_mask (tril causal) -- reconstructed analytically with the window
    const float* q_w    = (const float*)d_in[3];
    const float* kv_w   = (const float*)d_in[4];
    const float* out_w  = (const float*)d_in[5];
    float* out = (float*)d_out;

    const int M = BB * TT;  // 4096 token rows

    char* ws = (char*)d_ws;
    bf16_t* qb  = (bf16_t*)ws; ws += (size_t)M * NH  * HD * sizeof(bf16_t);
    bf16_t* kb  = (bf16_t*)ws; ws += (size_t)M * KVH * HD * sizeof(bf16_t);
    bf16_t* vb  = (bf16_t*)ws; ws += (size_t)M * KVH * HD * sizeof(bf16_t);
    bf16_t* enc = (bf16_t*)ws; ws += (size_t)M * NH  * HD * sizeof(bf16_t);

    // 1) projections (q, k, v) -> bf16
    proj_gemm_kernel<<<dim3(M / 16, HD / 64, NH),  128, 0, stream>>>(x, q_w,  qb, M, DD, NH);
    proj_gemm_kernel<<<dim3(M / 16, HD / 64, KVH), 128, 0, stream>>>(x, kv_w, kb, M, DD, KVH);
    proj_gemm_kernel<<<dim3(M / 16, HD / 64, KVH), 128, 0, stream>>>(
        x, kv_w + (size_t)KVH * DD * HD, vb, M, DD, KVH);

    // 2) RoPE in place (q also scaled)
    {
        int total = BB * TT * NH * (HD / 2) + BB * TT * KVH * (HD / 2);
        rope_kernel<<<(total + 255) / 256, 256, 0, stream>>>(qb, kb, segpos);
    }

    // 3) sliding-window flash attention
    attn_kernel<<<dim3(TT / 64, NH, BB), 128, 0, stream>>>(qb, kb, vb, enc);

    // 4) output projection -> f32 d_out
    outproj_kernel<<<dim3(M / 16, DD / 64), 128, 0, stream>>>(enc, out_w, out, M, NH * HD, DD);
}
